// MoMEAttentionAdaptor_66305705116280
// MI455X (gfx1250) — compile-verified, matrix operand order checked by reading, not run on hardware
//
#include <hip/hip_runtime.h>
#include <hip/hip_bf16.h>
#include <math.h>
#include <stdint.h>

typedef __attribute__((ext_vector_type(16))) _Float16 v16h;
typedef __attribute__((ext_vector_type(8)))  float    v8f;
typedef __attribute__((ext_vector_type(4)))  unsigned int u32x4;
typedef __attribute__((ext_vector_type(8)))  unsigned int u32x8;

namespace {
constexpr int kB   = 4;
constexpr int kS   = 2048;
constexpr int kH   = 2048;
constexpr int kNH  = 16;
constexpr int kHD  = 128;
constexpr int kBS  = kB * kS;      // 8192
constexpr int kR   = 64;
constexpr int kD   = 512;          // IDX_DIM
constexpr int kN   = 65536;        // N_INDEX
}

// ---------------------------------------------------------------------------
// WMMA operand helpers (CDNA5 16x16x32 f16, wave32).
// A (16x32): lane<16 -> row=lane, K elems {0..7,16..23}; lane>=16 -> same row,
//            K elems {8..15,24..31}.  element e <-> K = (e<8?e:e+8)+8*half.
// B (32x16): same element->K map, lane%16 -> column N (mirrors A).
// C/D (16x16 f32, v8f): VGPR i -> row i+8*half, col = lane%16.
// ---------------------------------------------------------------------------
__device__ __forceinline__ int kmap(int e, int half) {
  return (e < 8 ? e : e + 8) + half * 8;
}

__device__ __forceinline__ v8f wmma_f16(v16h a, v16h b, v8f c) {
  return __builtin_amdgcn_wmma_f32_16x16x32_f16(false, a, false, b, (short)0, c,
                                                false, false);
}

__device__ __forceinline__ v16h load_a(const _Float16* __restrict__ A, int lda,
                                       int r0, int k0, int lr, int half) {
  v16h a;
  const _Float16* p = A + (size_t)(r0 + lr) * lda + k0;
#pragma unroll
  for (int e = 0; e < 16; ++e) a[e] = p[kmap(e, half)];
  return a;
}

// B operand for C = A * B^T, B row-major [N, K] (i.e. x @ W.T with W=[out,in])
__device__ __forceinline__ v16h load_bt(const _Float16* __restrict__ Bm, int ldb,
                                        int n0, int k0, int lr, int half) {
  v16h b;
  const _Float16* p = Bm + (size_t)(n0 + lr) * ldb + k0;
#pragma unroll
  for (int e = 0; e < 16; ++e) b[e] = p[kmap(e, half)];
  return b;
}

// ---------------------------------------------------------------------------
// Tensor Data Mover: DMA a 2-D f16 tile (tile_rows x tile_k elems, row pitch
// ld_elems) from global into LDS at lds_addr (packed [tile_rows][tile_k]).
// D# per CDNA5 ISA 8.3/8.4: group0 = {count=1 | type=2 | lds | global addr},
// group1 = {data_size=2B, tensor dims == tile dims (tile always in-bounds),
// tile_dim0/1, dim0 stride}.  VADDR2/3 omitted -> 2-D tensor.
// Tracked on TENSORcnt; wave-level op (EXEC ignored); in-order completion.
// ---------------------------------------------------------------------------
__device__ __forceinline__ void tdm_load_tile_f16(unsigned int lds_addr,
                                                  const void* gptr,
                                                  unsigned int tile_k,
                                                  unsigned int tile_rows,
                                                  unsigned int ld_elems) {
  unsigned long long ga = (unsigned long long)(uintptr_t)gptr;
  u32x4 g0;
  g0[0] = 1u;                                              // count=1, user mode
  g0[1] = lds_addr;                                        // LDS byte address
  g0[2] = (unsigned int)ga;                                // global_addr[31:0]
  g0[3] = (unsigned int)((ga >> 32) & 0x1ffffffull) | (2u << 30);  // type=2
  u32x8 g1;
  g1[0] = 1u << 16;                                        // data_size=1 (2B)
  g1[1] = (tile_k & 0xffffu) << 16;                        // tensor_dim0 lo
  g1[2] = (tile_k >> 16) | ((tile_rows & 0xffffu) << 16);  // dim0 hi | dim1 lo
  g1[3] = (tile_rows >> 16) | ((tile_k & 0xffffu) << 16);  // dim1 hi | tile_dim0
  g1[4] = tile_rows & 0xffffu;                             // tile_dim1, tile_dim2=0
  g1[5] = ld_elems;                                        // dim0_stride[31:0]
  g1[6] = 0u;                                              // stride hi | dim1_stride lo
  g1[7] = 0u;
  asm volatile("tensor_load_to_lds %0, %1" :: "s"(g0), "s"(g1) : "memory");
}

// ---------------------------------------------------------------------------
__global__ void cvt_f32_f16(const float* __restrict__ src,
                            _Float16* __restrict__ dst, size_t n) {
  size_t i  = (size_t)blockIdx.x * blockDim.x + threadIdx.x;
  size_t st = (size_t)gridDim.x * blockDim.x;
  for (; i < n; i += st) dst[i] = (_Float16)src[i];
}

// Small GEMM (one 16x16 tile per wave) for skinny N=64 cases.
__global__ void gemm_tn_f16_small(const _Float16* __restrict__ A,
                                  const _Float16* __restrict__ Bm,
                                  _Float16* __restrict__ C,
                                  int Kd, int lda, int ldb, int ldc) {
  int lane = threadIdx.x & 31, lr = lane & 15, half = lane >> 4;
  int n0 = blockIdx.x * 16, m0 = blockIdx.y * 16;
  v8f acc = {};
  for (int k0 = 0; k0 < Kd; k0 += 32) {
    v16h a = load_a(A, lda, m0, k0, lr, half);
    v16h b = load_bt(Bm, ldb, n0, k0, lr, half);
    acc = wmma_f16(a, b, acc);
  }
#pragma unroll
  for (int i = 0; i < 8; ++i)
    C[(size_t)(m0 + i + 8 * half) * ldc + n0 + lr] = (_Float16)acc[i];
}

// ---------------------------------------------------------------------------
// Blocked GEMM: 256 threads = 8 waves as 2(M) x 4(N); WG tile 64x128, each
// wave 32x32 = 2x2 WMMA tiles.  A/B k-slices double-buffered in LDS via TDM:
// wave 0 issues the k+1 slice, then s_wait_tensorcnt(2) -- in-order TENSORcnt
// completion guarantees the current buffer's two DMAs are done while the two
// just issued stay in flight, overlapping DMA with WMMA.
// ---------------------------------------------------------------------------
__device__ __forceinline__ void gemm_big_core(const _Float16* __restrict__ A,
                                              const _Float16* __restrict__ Bm,
                                              int Kd, int lda, int ldb,
                                              _Float16* sA, _Float16* sB,
                                              v8f acc[2][2]) {
  int lane = threadIdx.x & 31, lr = lane & 15, half = lane >> 4;
  int w = threadIdx.x >> 5, wm = w & 1, wn = w >> 1;
  int m0 = blockIdx.y * 64, n0 = blockIdx.x * 128;
  int nk = Kd / 32;
  if (threadIdx.x < 32) {
    tdm_load_tile_f16((unsigned int)(uintptr_t)sA,
                      A + (size_t)m0 * lda, 32u, 64u, (unsigned int)lda);
    tdm_load_tile_f16((unsigned int)(uintptr_t)sB,
                      Bm + (size_t)n0 * ldb, 32u, 128u, (unsigned int)ldb);
  }
  for (int t = 0; t < nk; ++t) {
    int buf = t & 1;
    _Float16* cA = sA + buf * (64 * 32);
    _Float16* cB = sB + buf * (128 * 32);
    if (threadIdx.x < 32) {
      if (t + 1 < nk) {
        int k1 = (t + 1) * 32;
        _Float16* nA = sA + (buf ^ 1) * (64 * 32);
        _Float16* nB = sB + (buf ^ 1) * (128 * 32);
        tdm_load_tile_f16((unsigned int)(uintptr_t)nA,
                          A + (size_t)m0 * lda + k1, 32u, 64u,
                          (unsigned int)lda);
        tdm_load_tile_f16((unsigned int)(uintptr_t)nB,
                          Bm + (size_t)n0 * ldb + k1, 32u, 128u,
                          (unsigned int)ldb);
        __builtin_amdgcn_s_wait_tensorcnt(2);  // current buffer complete
      } else {
        __builtin_amdgcn_s_wait_tensorcnt(0);
      }
    }
    __syncthreads();
    v16h af[2], bf[2];
#pragma unroll
    for (int mi = 0; mi < 2; ++mi)
#pragma unroll
      for (int e = 0; e < 16; ++e)
        af[mi][e] = cA[(wm * 32 + mi * 16 + lr) * 32 + kmap(e, half)];
#pragma unroll
    for (int ni = 0; ni < 2; ++ni)
#pragma unroll
      for (int e = 0; e < 16; ++e)
        bf[ni][e] = cB[(wn * 32 + ni * 16 + lr) * 32 + kmap(e, half)];
#pragma unroll
    for (int mi = 0; mi < 2; ++mi)
#pragma unroll
      for (int ni = 0; ni < 2; ++ni)
        acc[mi][ni] = wmma_f16(af[mi], bf[ni], acc[mi][ni]);
    __syncthreads();  // reads of this buffer done before it is re-targeted
  }
}

__global__ void gemm_tn_f16_big(const _Float16* __restrict__ A,
                                const _Float16* __restrict__ Bm,
                                _Float16* __restrict__ C,
                                int Kd, int lda, int ldb, int ldc) {
  __shared__ _Float16 sA[2][64 * 32];
  __shared__ _Float16 sB[2][128 * 32];
  v8f acc[2][2] = {};
  gemm_big_core(A, Bm, Kd, lda, ldb, &sA[0][0], &sB[0][0], acc);
  int lane = threadIdx.x & 31, lr = lane & 15, half = lane >> 4;
  int w = threadIdx.x >> 5, wm = w & 1, wn = w >> 1;
  int m0 = blockIdx.y * 64, n0 = blockIdx.x * 128;
#pragma unroll
  for (int mi = 0; mi < 2; ++mi)
#pragma unroll
    for (int ni = 0; ni < 2; ++ni)
#pragma unroll
      for (int i = 0; i < 8; ++i)
        C[(size_t)(m0 + wm * 32 + mi * 16 + i + 8 * half) * ldc +
          n0 + wn * 32 + ni * 16 + lr] = (_Float16)acc[mi][ni][i];
}

__global__ void gemm_tn_f32_big(const _Float16* __restrict__ A,
                                const _Float16* __restrict__ Bm,
                                float* __restrict__ C,
                                int Kd, int lda, int ldb, int ldc,
                                float alpha, int accumulate) {
  __shared__ _Float16 sA[2][64 * 32];
  __shared__ _Float16 sB[2][128 * 32];
  v8f acc[2][2] = {};
  gemm_big_core(A, Bm, Kd, lda, ldb, &sA[0][0], &sB[0][0], acc);
  int lane = threadIdx.x & 31, lr = lane & 15, half = lane >> 4;
  int w = threadIdx.x >> 5, wm = w & 1, wn = w >> 1;
  int m0 = blockIdx.y * 64, n0 = blockIdx.x * 128;
#pragma unroll
  for (int mi = 0; mi < 2; ++mi)
#pragma unroll
    for (int ni = 0; ni < 2; ++ni)
#pragma unroll
      for (int i = 0; i < 8; ++i) {
        float* p = C + (size_t)(m0 + wm * 32 + mi * 16 + i + 8 * half) * ldc +
                   n0 + wn * 32 + ni * 16 + lr;
        float v = alpha * acc[mi][ni][i];
        *p = accumulate ? (*p + v) : v;
      }
}

// ---------------------------------------------------------------------------
// Base causal MHA, flash style.  grid = (S/16, B*NH), block = 32 (1 wave).
// Q/K/V are f16 [B*S, H], head h at columns h*128..h*128+127.
// K/V 32-key slices staged via TDM each key block.
// ---------------------------------------------------------------------------
__global__ void base_attn_kernel(const _Float16* __restrict__ Q,
                                 const _Float16* __restrict__ K,
                                 const _Float16* __restrict__ V,
                                 _Float16* __restrict__ O) {
  __shared__ _Float16 s_k[32 * 128];
  __shared__ _Float16 s_v[32 * 128];
  __shared__ _Float16 s_p[16 * 32];
  int lane = threadIdx.x & 31, lr = lane & 15, half = lane >> 4;
  int q0 = blockIdx.x * 16;
  int bh = blockIdx.y, b = bh >> 4, h = bh & 15;
  size_t boff = (size_t)b * kS * kH;
  const _Float16* Qb = Q + boff + h * kHD;
  const _Float16* Kb = K + boff + h * kHD;
  const _Float16* Vb = V + boff + h * kHD;
  const float scale = 0.08838834764831845f;  // 1/sqrt(128)
  const float NEGINF = -__builtin_inff();

  v16h qa[4];
#pragma unroll
  for (int ks = 0; ks < 4; ++ks) qa[ks] = load_a(Qb, kH, q0, ks * 32, lr, half);

  v8f o[8];
  float m_i[8], l_i[8], al_i[8];
#pragma unroll
  for (int f = 0; f < 8; ++f) o[f] = (v8f){};
#pragma unroll
  for (int i = 0; i < 8; ++i) { m_i[i] = NEGINF; l_i[i] = 0.f; }

  int kbmax = (q0 + 15) / 32;
  for (int kb = 0; kb <= kbmax; ++kb) {
    int kbase = kb * 32;
    // WAR: finish previous iteration's LDS reads before the TDM rewrites them
    asm volatile("s_wait_dscnt 0x0" ::: "memory");
    tdm_load_tile_f16((unsigned int)(uintptr_t)s_k,
                      Kb + (size_t)kbase * kH, 128u, 32u, (unsigned int)kH);
    tdm_load_tile_f16((unsigned int)(uintptr_t)s_v,
                      Vb + (size_t)kbase * kH, 128u, 32u, (unsigned int)kH);
    __builtin_amdgcn_s_wait_tensorcnt(0);

    v8f s0 = {}, s1 = {};
#pragma unroll
    for (int ks = 0; ks < 4; ++ks) {
      v16h b0, b1;
#pragma unroll
      for (int e = 0; e < 16; ++e) {
        b0[e] = s_k[lr * 128 + ks * 32 + kmap(e, half)];
        b1[e] = s_k[(16 + lr) * 128 + ks * 32 + kmap(e, half)];
      }
      s0 = wmma_f16(qa[ks], b0, s0);
      s1 = wmma_f16(qa[ks], b1, s1);
    }
#pragma unroll
    for (int i = 0; i < 8; ++i) {
      int qi = q0 + i + 8 * half;
      s0[i] = (kbase + lr <= qi) ? s0[i] * scale : NEGINF;
      s1[i] = (kbase + 16 + lr <= qi) ? s1[i] * scale : NEGINF;
    }
    // online softmax: row reductions across 16-lane halves
#pragma unroll
    for (int i = 0; i < 8; ++i) {
      float mx = fmaxf(s0[i], s1[i]);
      mx = fmaxf(mx, __shfl_xor(mx, 1, 32));
      mx = fmaxf(mx, __shfl_xor(mx, 2, 32));
      mx = fmaxf(mx, __shfl_xor(mx, 4, 32));
      mx = fmaxf(mx, __shfl_xor(mx, 8, 32));
      float mnew = fmaxf(m_i[i], mx);
      float al = __expf(m_i[i] - mnew);
      float p0 = __expf(s0[i] - mnew);
      float p1 = __expf(s1[i] - mnew);
      float rs = p0 + p1;
      rs += __shfl_xor(rs, 1, 32);
      rs += __shfl_xor(rs, 2, 32);
      rs += __shfl_xor(rs, 4, 32);
      rs += __shfl_xor(rs, 8, 32);
      l_i[i] = l_i[i] * al + rs;
      m_i[i] = mnew;
      al_i[i] = al;
      s0[i] = p0;
      s1[i] = p1;
    }
#pragma unroll
    for (int f = 0; f < 8; ++f)
#pragma unroll
      for (int i = 0; i < 8; ++i) o[f][i] *= al_i[i];
    // restage P: C-layout -> LDS -> A-layout (LDS in-order within wave)
#pragma unroll
    for (int i = 0; i < 8; ++i) {
      int r = i + 8 * half;
      s_p[r * 32 + lr] = (_Float16)s0[i];
      s_p[r * 32 + 16 + lr] = (_Float16)s1[i];
    }
    asm volatile("s_wait_dscnt 0x0" ::: "memory");
    v16h pa;
#pragma unroll
    for (int e = 0; e < 16; ++e) pa[e] = s_p[lr * 32 + kmap(e, half)];
    // P @ V over 8 feature tiles, V fragments from LDS tile
#pragma unroll
    for (int f = 0; f < 8; ++f) {
      v16h vb;
#pragma unroll
      for (int e = 0; e < 16; ++e)
        vb[e] = s_v[kmap(e, half) * 128 + f * 16 + lr];
      o[f] = wmma_f16(pa, vb, o[f]);
    }
  }
#pragma unroll
  for (int f = 0; f < 8; ++f)
#pragma unroll
    for (int i = 0; i < 8; ++i) {
      int r = q0 + i + 8 * half;
      O[boff + (size_t)r * kH + h * kHD + f * 16 + lr] =
          (_Float16)(o[f][i] / l_i[i]);
    }
}

// ---------------------------------------------------------------------------
// sim = query[8192,512] @ index_keys[65536,512]^T with per-row top-4.
// block = 256 (8 waves) handles 16 query rows (TDM-staged in LDS once);
// waves tile 128 keys per chunk; keys stream stays L2-resident (64MB < 192MB).
// Next chunk's key rows prefetched (global_prefetch_b8).
// ---------------------------------------------------------------------------
__global__ void sim_topk_kernel(const _Float16* __restrict__ Qr,
                                const _Float16* __restrict__ Keys,
                                int* __restrict__ top_idx) {
  __shared__ _Float16 s_q[16 * 512];
  __shared__ float s_sc[16 * 128];
  __shared__ float s_mv[16 * 64];
  __shared__ int s_mi[16 * 64];
  int tid = threadIdx.x;
  int wave = tid >> 5, lane = tid & 31, lr = lane & 15, half = lane >> 4;
  int m0 = blockIdx.x * 16;
  const float NEGINF = -__builtin_inff();
  float bv[4] = {NEGINF, NEGINF, NEGINF, NEGINF};
  int bi[4] = {0, 0, 0, 0};
  int r = tid >> 4, c0 = (tid & 15) * 8;

  if (tid < 32) {
    tdm_load_tile_f16((unsigned int)(uintptr_t)s_q,
                      Qr + (size_t)m0 * kD, (unsigned int)kD, 16u,
                      (unsigned int)kD);
    __builtin_amdgcn_s_wait_tensorcnt(0);
  }
  __syncthreads();

  for (int chunk = 0; chunk < kN; chunk += 128) {
    int n0 = chunk + wave * 16;
    if (chunk + 128 < kN)  // pull next chunk's rows toward L0
      __builtin_prefetch(Keys + (size_t)(n0 + 128 + lr) * kD + half * 256, 0, 0);
    v8f acc = {};
#pragma unroll
    for (int ks = 0; ks < 16; ++ks) {
      v16h a, b;
#pragma unroll
      for (int e = 0; e < 16; ++e)
        a[e] = s_q[lr * kD + ks * 32 + kmap(e, half)];
      b = load_bt(Keys, kD, n0, ks * 32, lr, half);
      acc = wmma_f16(a, b, acc);
    }
#pragma unroll
    for (int i = 0; i < 8; ++i)
      s_sc[(i + 8 * half) * 128 + wave * 16 + lr] = acc[i];
    __syncthreads();
#pragma unroll
    for (int j = 0; j < 8; ++j) {
      float v = s_sc[r * 128 + c0 + j];
      int gi = chunk + c0 + j;
      if (v > bv[3]) {
        if (v > bv[0]) {
          bv[3]=bv[2]; bi[3]=bi[2]; bv[2]=bv[1]; bi[2]=bi[1];
          bv[1]=bv[0]; bi[1]=bi[0]; bv[0]=v; bi[0]=gi;
        } else if (v > bv[1]) {
          bv[3]=bv[2]; bi[3]=bi[2]; bv[2]=bv[1]; bi[2]=bi[1];
          bv[1]=v; bi[1]=gi;
        } else if (v > bv[2]) {
          bv[3]=bv[2]; bi[3]=bi[2]; bv[2]=v; bi[2]=gi;
        } else { bv[3]=v; bi[3]=gi; }
      }
    }
    __syncthreads();
  }
  int sl = tid & 15;
#pragma unroll
  for (int j = 0; j < 4; ++j) {
    s_mv[r * 64 + sl * 4 + j] = bv[j];
    s_mi[r * 64 + sl * 4 + j] = bi[j];
  }
  __syncthreads();
  if (tid < 16) {
    for (int j = 0; j < 4; ++j) {
      float best = -__builtin_inff();
      int bidx = 0, bpos = 0;
      for (int t = 0; t < 64; ++t) {
        float v = s_mv[tid * 64 + t];
        if (v > best) { best = v; bidx = s_mi[tid * 64 + t]; bpos = t; }
      }
      s_mv[tid * 64 + bpos] = -__builtin_inff();
      top_idx[(size_t)(m0 + tid) * 4 + j] = bidx;
    }
  }
}

// ---------------------------------------------------------------------------
// mome attention, flash style with gathered K/V rows, LDS f32 O accumulator
// (16x512 doesn't fit VGPRs), Q tile TDM-staged once.
// grid = (S/16, B), block = 32.  Only key slots t <= i matter (t < 2048).
// ---------------------------------------------------------------------------
__global__ void mome_attn_kernel(const _Float16* __restrict__ Qr,
                                 const _Float16* __restrict__ IK,
                                 const _Float16* __restrict__ IV,
                                 const int* __restrict__ top_idx,
                                 _Float16* __restrict__ Out) {
  __shared__ float s_o[16 * 512];     // 32 KB O accumulator
  __shared__ _Float16 s_q[16 * 512];  // 16 KB staged Q tile
  __shared__ _Float16 s_p[16 * 32];
  __shared__ float s_bc[16];
  int lane = threadIdx.x & 31, lr = lane & 15, half = lane >> 4;
  int q0 = blockIdx.x * 16, b = blockIdx.y;
  const _Float16* Qb = Qr + (size_t)b * kS * kD;
  const int* idxb = top_idx + (size_t)b * (kS * 4);
  const float scale = 0.044194173824159216f;  // 1/sqrt(512)
  const float NEGINF = -__builtin_inff();

  tdm_load_tile_f16((unsigned int)(uintptr_t)s_q,
                    Qb + (size_t)q0 * kD, (unsigned int)kD, 16u,
                    (unsigned int)kD);
  for (int t = lane; t < 16 * 512; t += 32) s_o[t] = 0.f;
  __builtin_amdgcn_s_wait_tensorcnt(0);

  float m_i[8], l_i[8], al_i[8];
#pragma unroll
  for (int i = 0; i < 8; ++i) { m_i[i] = NEGINF; l_i[i] = 0.f; }

  int kbmax = (q0 + 15) / 32;
  for (int kb = 0; kb <= kbmax; ++kb) {
    int kbase = kb * 32;
    int gr0 = idxb[kbase + lr];
    int gr1 = idxb[kbase + 16 + lr];
    const _Float16* kr0 = IK + (size_t)gr0 * kD;
    const _Float16* kr1 = IK + (size_t)gr1 * kD;
    v8f s0 = {}, s1 = {};
#pragma unroll
    for (int ks = 0; ks < 16; ++ks) {
      int k0 = ks * 32;
      v16h a, b0, b1;
#pragma unroll
      for (int e = 0; e < 16; ++e) {
        a[e]  = s_q[lr * kD + k0 + kmap(e, half)];
        b0[e] = kr0[k0 + kmap(e, half)];
        b1[e] = kr1[k0 + kmap(e, half)];
      }
      s0 = wmma_f16(a, b0, s0);
      s1 = wmma_f16(a, b1, s1);
    }
#pragma unroll
    for (int i = 0; i < 8; ++i) {
      int qi = q0 + i + 8 * half;
      s0[i] = (kbase + lr <= qi) ? s0[i] * scale : NEGINF;
      s1[i] = (kbase + 16 + lr <= qi) ? s1[i] * scale : NEGINF;
    }
#pragma unroll
    for (int i = 0; i < 8; ++i) {
      float mx = fmaxf(s0[i], s1[i]);
      mx = fmaxf(mx, __shfl_xor(mx, 1, 32));
      mx = fmaxf(mx, __shfl_xor(mx, 2, 32));
      mx = fmaxf(mx, __shfl_xor(mx, 4, 32));
      mx = fmaxf(mx, __shfl_xor(mx, 8, 32));
      float mnew = fmaxf(m_i[i], mx);
      float al = __expf(m_i[i] - mnew);
      float p0 = __expf(s0[i] - mnew);
      float p1 = __expf(s1[i] - mnew);
      float rs = p0 + p1;
      rs += __shfl_xor(rs, 1, 32);
      rs += __shfl_xor(rs, 2, 32);
      rs += __shfl_xor(rs, 4, 32);
      rs += __shfl_xor(rs, 8, 32);
      l_i[i] = l_i[i] * al + rs;
      m_i[i] = mnew;
      al_i[i] = al;
      s0[i] = p0;
      s1[i] = p1;
    }
    // broadcast per-row alpha via LDS, rescale O accumulator
    if ((lane & 15) == 0) {
#pragma unroll
      for (int i = 0; i < 8; ++i) s_bc[i + 8 * half] = al_i[i];
    }
    asm volatile("s_wait_dscnt 0x0" ::: "memory");
    for (int t = lane; t < 16 * 512; t += 32) s_o[t] *= s_bc[t >> 9];
    // restage P in A layout
#pragma unroll
    for (int i = 0; i < 8; ++i) {
      int r = i + 8 * half;
      s_p[r * 32 + lr] = (_Float16)s0[i];
      s_p[r * 32 + 16 + lr] = (_Float16)s1[i];
    }
    asm volatile("s_wait_dscnt 0x0" ::: "memory");
    v16h pa;
#pragma unroll
    for (int e = 0; e < 16; ++e) pa[e] = s_p[lr * 32 + kmap(e, half)];
    // gathered V row ids for this lane's B elements (same for all f tiles)
    int vr[16];
#pragma unroll
    for (int e = 0; e < 16; ++e) vr[e] = idxb[kbase + kmap(e, half)];
    for (int f = 0; f < 32; ++f) {
      v16h vb;
#pragma unroll
      for (int e = 0; e < 16; ++e) vb[e] = IV[(size_t)vr[e] * kD + f * 16 + lr];
      v8f z = {};
      v8f t = wmma_f16(pa, vb, z);
#pragma unroll
      for (int i = 0; i < 8; ++i)
        s_o[(i + 8 * half) * 512 + f * 16 + lr] += t[i];
    }
  }
  // normalize and store
  if ((lane & 15) == 0) {
#pragma unroll
    for (int i = 0; i < 8; ++i) s_bc[i + 8 * half] = l_i[i];
  }
  asm volatile("s_wait_dscnt 0x0" ::: "memory");
  for (int f = 0; f < 32; ++f)
#pragma unroll
    for (int i = 0; i < 8; ++i) {
      int r = i + 8 * half;
      float v = s_o[r * 512 + f * 16 + lr] / s_bc[r];
      Out[((size_t)b * kS + q0 + r) * kD + f * 16 + lr] = (_Float16)v;
    }
}

// ---------------------------------------------------------------------------
extern "C" void kernel_launch(void* const* d_in, const int* in_sizes, int n_in,
                              void* d_out, int out_size, void* d_ws,
                              size_t ws_size, hipStream_t stream) {
  (void)in_sizes; (void)n_in; (void)out_size; (void)ws_size;
  const float* hs   = (const float*)d_in[0];
  const float* Wq   = (const float*)d_in[1];
  const float* Wk   = (const float*)d_in[2];
  const float* Wv   = (const float*)d_in[3];
  const float* Wo   = (const float*)d_in[4];
  const float* qinw = (const float*)d_in[5];
  const float* qouw = (const float*)d_in[6];
  const float* vinw = (const float*)d_in[7];
  const float* vouw = (const float*)d_in[8];
  const float* ikey = (const float*)d_in[9];
  const float* ival = (const float*)d_in[10];
  float* out = (float*)d_out;

  char* ws = (char*)d_ws;
  size_t off = 0;
  auto alloc = [&](size_t bytes) -> void* {
    void* p = ws + off;
    off += (bytes + 255) & ~(size_t)255;
    return p;
  };
  _Float16* hs_h   = (_Float16*)alloc((size_t)kBS * kH * 2);
  _Float16* Wq_h   = (_Float16*)alloc((size_t)kH * kH * 2);
  _Float16* Wk_h   = (_Float16*)alloc((size_t)kH * kH * 2);
  _Float16* Wv_h   = (_Float16*)alloc((size_t)kH * kH * 2);
  _Float16* Wo_h   = (_Float16*)alloc((size_t)kH * kH * 2);
  _Float16* qin_h  = (_Float16*)alloc((size_t)kR * kH * 2);
  _Float16* qout_h = (_Float16*)alloc((size_t)kD * kR * 2);
  _Float16* vin_h  = (_Float16*)alloc((size_t)kR * kD * 2);
  _Float16* vout_h = (_Float16*)alloc((size_t)kH * kR * 2);
  _Float16* IK_h   = (_Float16*)alloc((size_t)kN * kD * 2);
  _Float16* IV_h   = (_Float16*)alloc((size_t)kN * kD * 2);
  _Float16* Qh     = (_Float16*)alloc((size_t)kBS * kH * 2);
  _Float16* Kh     = (_Float16*)alloc((size_t)kBS * kH * 2);
  _Float16* Vh     = (_Float16*)alloc((size_t)kBS * kH * 2);
  _Float16* AOh    = (_Float16*)alloc((size_t)kBS * kH * 2);
  _Float16* t64    = (_Float16*)alloc((size_t)kBS * kR * 2);
  _Float16* qry_h  = (_Float16*)alloc((size_t)kBS * kD * 2);
  int*      topi   = (int*)alloc((size_t)kBS * 4 * 4);
  _Float16* mome_h = (_Float16*)alloc((size_t)kBS * kD * 2);
  _Float16* t64b   = (_Float16*)alloc((size_t)kBS * kR * 2);

  auto cvt = [&](const float* s, _Float16* d, size_t n) {
    int grid = (int)((n + 255) / 256);
    if (grid > 4096) grid = 4096;
    cvt_f32_f16<<<grid, 256, 0, stream>>>(s, d, n);
  };
  cvt(hs, hs_h, (size_t)kBS * kH);
  cvt(Wq, Wq_h, (size_t)kH * kH);
  cvt(Wk, Wk_h, (size_t)kH * kH);
  cvt(Wv, Wv_h, (size_t)kH * kH);
  cvt(Wo, Wo_h, (size_t)kH * kH);
  cvt(qinw, qin_h, (size_t)kR * kH);
  cvt(qouw, qout_h, (size_t)kD * kR);
  cvt(vinw, vin_h, (size_t)kR * kD);
  cvt(vouw, vout_h, (size_t)kH * kR);
  cvt(ikey, IK_h, (size_t)kN * kD);
  cvt(ival, IV_h, (size_t)kN * kD);

  // base QKV projections (TDM double-buffered blocked GEMM, WG tile 64x128)
  dim3 gBig(kH / 128, kBS / 64);
  gemm_tn_f16_big<<<gBig, 256, 0, stream>>>(hs_h, Wq_h, Qh, kH, kH, kH, kH);
  gemm_tn_f16_big<<<gBig, 256, 0, stream>>>(hs_h, Wk_h, Kh, kH, kH, kH, kH);
  gemm_tn_f16_big<<<gBig, 256, 0, stream>>>(hs_h, Wv_h, Vh, kH, kH, kH, kH);

  // causal MHA, then O projection (beta=0) directly into d_out
  base_attn_kernel<<<dim3(kS / 16, kB * kNH), 32, 0, stream>>>(Qh, Kh, Vh, AOh);
  gemm_tn_f32_big<<<gBig, 256, 0, stream>>>(AOh, Wo_h, out, kH, kH, kH, kH,
                                            1.0f, 0);

  // adapter query: hs @ q_in_w.T -> [8192,64]; @ q_out_w.T -> [8192,512]
  gemm_tn_f16_small<<<dim3(kR / 16, kBS / 16), 32, 0, stream>>>(
      hs_h, qin_h, t64, kH, kH, kH, kR);
  gemm_tn_f16_big<<<dim3(kD / 128, kBS / 64), 256, 0, stream>>>(
      t64, qout_h, qry_h, kR, kR, kR, kD);

  // top-4 retrieval over 65536 index keys (L2-resident f16 stream)
  sim_topk_kernel<<<kBS / 16, 256, 0, stream>>>(qry_h, IK_h, topi);

  // mome causal attention over gathered keys/values
  mome_attn_kernel<<<dim3(kS / 16, kB), 32, 0, stream>>>(
      qry_h, IK_h, IV_h, topi, mome_h);

  // mome projection: @ v_in_w.T -> [8192,64]; alpha=R into d_out (beta=1)
  gemm_tn_f16_small<<<dim3(kR / 16, kBS / 16), 32, 0, stream>>>(
      mome_h, vin_h, t64b, kD, kD, kD, kR);
  gemm_tn_f32_big<<<gBig, 256, 0, stream>>>(t64b, vout_h, out, kR, kR, kR, kH,
                                            (float)kR, 1);
}
